// GCNSE_50130858279707
// MI455X (gfx1250) — matrix operand order repaired and checked by reading, not compiled, not run on hardware
//
#include <hip/hip_runtime.h>
#include <cstddef>

typedef __attribute__((ext_vector_type(16))) _Float16 v16h;
typedef __attribute__((ext_vector_type(2)))  _Float16 v2h;
typedef __attribute__((ext_vector_type(8)))  float    v8f;

// Problem dims (fixed by reference setup_inputs)
constexpr int T   = 10;
constexpr int Bb  = 4;
constexpr int N   = 1000;
constexpr int BN  = Bb * N;      // 4000
constexpr int D   = 16;
constexpr int H   = 64;
constexpr int O   = 32;
constexpr int TSQ = T / 2;

// Prop-GEMM tiling: 8 waves * 32 = 256 threads; wave w owns a 16-row M-tile;
// block tile = 128 rows x 64 cols (full H); K-step = 32; double-buffered LDS.
constexpr int MT = 128;
constexpr int KT = 32;

// Workspace layout (bytes). Total ~26.1 MB.
constexpr size_t SZ_TBN  = (size_t)T * BN * sizeof(float);          // 160000
constexpr size_t OFF_MASK   = 0;
constexpr size_t OFF_DINV   = OFF_MASK   + SZ_TBN;
constexpr size_t OFF_COLSUM = OFF_DINV   + SZ_TBN;
constexpr size_t OFF_C      = OFF_COLSUM + SZ_TBN;                  // T floats
constexpr size_t OFF_S      = OFF_C + 512;
constexpr size_t OFF_Y      = OFF_S + 512;                          // T*BN*H f16
constexpr size_t OFF_H1     = OFF_Y  + (size_t)T * BN * H * 2;
constexpr size_t OFF_H2     = OFF_H1 + (size_t)T * BN * H * sizeof(float);

// ---------------- mask build + colsum zero ----------------
__global__ void k_mask(const int* __restrict__ ego, float* __restrict__ mask,
                       float* __restrict__ colsum) {
    int idx = blockIdx.x * 256 + threadIdx.x;
    if (idx >= T * BN) return;
    int t = idx / BN, bn = idx % BN;
    int b = bn / N,  n  = bn % N;
    mask[idx]   = (float)ego[((size_t)b * T + t) * N + n];
    colsum[idx] = 0.0f;
}

// ---------------- masked column sums of adjacency ----------------
constexpr int JSPLIT = 8;
constexpr int JCH    = (BN + JSPLIT - 1) / JSPLIT;   // 500
__global__ void k_colsum(const int* __restrict__ adj, const float* __restrict__ mask,
                         float* __restrict__ colsum) {
    int t = blockIdx.y;
    int i = blockIdx.x * 256 + threadIdx.x;
    if (i >= BN) return;
    int j0 = blockIdx.z * JCH;
    int j1 = j0 + JCH; if (j1 > BN) j1 = BN;
    const int*   a  = adj  + (size_t)t * BN * BN + i;
    const float* mk = mask + (size_t)t * BN;
    float acc = 0.0f;
    for (int j = j0; j < j1; ++j)
        acc += (float)a[(size_t)j * BN] * mk[j];
    atomicAdd(&colsum[(size_t)t * BN + i], acc);
}

__global__ void k_dinv(const float* __restrict__ mask, const float* __restrict__ colsum,
                       float* __restrict__ dinv) {
    int idx = blockIdx.x * 256 + threadIdx.x;
    if (idx >= T * BN) return;
    dinv[idx] = (mask[idx] > 0.5f) ? rsqrtf(colsum[idx] + 1.0f) : 0.0f;
}

// ---------------- Y1 = f16( dinv * (x @ W1) ) ----------------
__global__ void k_y1(const float* __restrict__ x, const float* __restrict__ W1,
                     const float* __restrict__ dinv, _Float16* __restrict__ Y) {
    int idx = blockIdx.x * 256 + threadIdx.x;
    if (idx >= T * BN * H) return;
    int h = idx % H, tj = idx / H;
    const float* xr = x + (size_t)tj * D;
    float acc = 0.0f;
    #pragma unroll
    for (int d = 0; d < D; ++d) acc += xr[d] * W1[d * H + h];
    Y[idx] = (_Float16)(dinv[tj] * acc);
}

// ---------------- Y2 = f16( dinv * (h1 @ W2) ) ----------------
__global__ void k_y2(const float* __restrict__ h1, const float* __restrict__ W2,
                     const float* __restrict__ dinv, _Float16* __restrict__ Y) {
    int idx = blockIdx.x * 256 + threadIdx.x;
    if (idx >= T * BN * H) return;
    int h = idx % H, tj = idx / H;
    const float* hr = h1 + (size_t)tj * H;
    float acc = 0.0f;
    #pragma unroll 8
    for (int k = 0; k < H; ++k) acc += hr[k] * W2[k * H + h];
    Y[idx] = (_Float16)(dinv[tj] * acc);
}

// ---------------- WMMA propagation: out = epi( dinv_i * (M @ Y) ) ----------------
// M[i,j] = adj[t,j,i] + (i==j); LDS tiles held in WMMA *fragment order* so each
// lane reads its whole fragment as one contiguous 32B block (2x ds_load_b128).
// Double-buffered: next tile is staged while the current one feeds the WMMAs.
// EPI==0: h1 = relu(v + b1);  EPI==1: h2 = (v + b2) * mask
template <int EPI>
__global__ void __launch_bounds__(256) k_prop(const int* __restrict__ adj,
                                              const _Float16* __restrict__ Y,
                                              const float* __restrict__ dinv,
                                              const float* __restrict__ mask,
                                              const float* __restrict__ bias,
                                              float* __restrict__ out) {
    // [2 buffers][8 wave-tiles][32 lanes][16 f16]  A fragments, 16KB
    __shared__ alignas(32) _Float16 lAf[2][8 * 32 * 16];
    // [2 buffers][4 n-tiles ][32 lanes][16 f16]  B fragments, 8KB
    __shared__ alignas(32) _Float16 lBf[2][4 * 32 * 16];

    const int t    = blockIdx.y;
    const int i0   = blockIdx.x * MT;
    const int tid  = threadIdx.x;
    const int lane = tid & 31;
    const int wave = tid >> 5;
    const int half = lane >> 4;
    const int l16  = lane & 15;

    // A staging decomposition: tid = sHalf*128 + sWt*16 + sL16
    const int  sL16  = tid & 15;
    const int  sWt   = (tid >> 4) & 7;
    const int  sHalf = tid >> 7;
    const int  sM    = tid & 127;                 // row-in-tile this thread stages
    const int  sLane = sHalf * 16 + sL16;         // destination fragment lane
    const int  gi    = i0 + sM;                   // global column index i
    const bool giOK  = gi < BN;

    // B staging decomposition: tid bits [3:0]=l16 [5:4]=nt [6]=half [7]=vhi
    const int bL16  = tid & 15;
    const int bNt   = (tid >> 4) & 3;
    const int bHalf = (tid >> 6) & 1;
    const int bVhi  = tid >> 7;
    const int bH    = bNt * 16 + bL16;
    const int bLane = bHalf * 16 + bL16;

    const int*      adjT = adj + (size_t)t * BN * BN;
    const _Float16* Yt   = Y   + (size_t)t * BN * H;

    // Per-thread stage-destination and fragment-read pointers, both buffers.
    _Float16* const stA0 = &lAf[0][(sWt * 32 + sLane) << 4];
    _Float16* const stA1 = &lAf[1][(sWt * 32 + sLane) << 4];
    _Float16* const stB0 = &lBf[0][(bNt * 32 + bLane) << 4];
    _Float16* const stB1 = &lBf[1][(bNt * 32 + bLane) << 4];
    const _Float16* const rdA0 = &lAf[0][(wave * 32 + lane) << 4];
    const _Float16* const rdA1 = &lAf[1][(wave * 32 + lane) << 4];
    const _Float16* const rdB0 = &lBf[0][lane << 4];
    const _Float16* const rdB1 = &lBf[1][lane << 4];

    // Stage A fragments for K-block starting at row jb (packed f16 pairs).
    // Per-ISA A layout: lane L -> M=L%16; VGPR v holds K pair
    // kb = (v<4 ? 2v : 8+2v) + 8*(L/16).
    auto stageA = [&](int jb, _Float16* dA) {
        const int* colp = adjT + (size_t)jb * BN + gi;
        #pragma unroll
        for (int v = 0; v < 8; ++v) {
            const int kb = ((v < 4) ? (2 * v) : (8 + 2 * v)) + 8 * sHalf;
            float f0 = 0.0f, f1 = 0.0f;
            if (giOK) {
                f0 = (float)colp[(size_t)kb * BN];
                f1 = (float)colp[(size_t)(kb + 1) * BN];
                if (jb + kb     == gi) f0 += 1.0f;
                if (jb + kb + 1 == gi) f1 += 1.0f;
            }
            v2h p; p[0] = (_Float16)f0; p[1] = (_Float16)f1;
            *(v2h*)&dA[2 * v] = p;
        }
    };
    // Stage B fragments. B layout: lane L -> N=L%16; VGPR v holds K pair
    // k = (v<4 ? 2v : 8+2v) + 8*(L/16).
    auto stageB = [&](int jb, _Float16* dB) {
        const _Float16* yp = Yt + (size_t)jb * H + bH;
        #pragma unroll
        for (int v2 = 0; v2 < 4; ++v2) {
            const int v = bVhi * 4 + v2;
            const int k = ((v < 4) ? (2 * v) : (8 + 2 * v)) + 8 * bHalf;
            v2h p;
            p[0] = yp[(size_t)k * H];
            p[1] = yp[(size_t)(k + 1) * H];
            *(v2h*)&dB[2 * v] = p;
        }
    };

    v8f acc[4] = {v8f{}, v8f{}, v8f{}, v8f{}};

    // Prologue: stage first K-block into buffer 0.
    stageA(0, stA0);
    stageB(0, stB0);
    __syncthreads();

    int p = 0;
    for (int j0 = 0; j0 < BN; j0 += KT) {
        // Kick off staging of the next K-block into the other buffer first so
        // its global loads overlap the WMMA chain below.
        const int jn = j0 + KT;
        if (jn < BN) {
            stageA(jn, p ? stA0 : stA1);
            stageB(jn, p ? stB0 : stB1);
        }

        // Consume current buffer: contiguous 32B fragment reads, then 4 WMMAs.
        const _Float16* rA = p ? rdA1 : rdA0;
        const _Float16* rB = p ? rdB1 : rdB0;
        v16h a  = *(const v16h*)rA;
        v16h b0 = *(const v16h*)(rB);
        v16h b1 = *(const v16h*)(rB + 1 * 32 * 16);
        v16h b2 = *(const v16h*)(rB + 2 * 32 * 16);
        v16h b3 = *(const v16h*)(rB + 3 * 32 * 16);
        acc[0] = __builtin_amdgcn_wmma_f32_16x16x32_f16(false, a, false, b0, (short)0, acc[0], false, false);
        acc[1] = __builtin_amdgcn_wmma_f32_16x16x32_f16(false, a, false, b1, (short)0, acc[1], false, false);
        acc[2] = __builtin_amdgcn_wmma_f32_16x16x32_f16(false, a, false, b2, (short)0, acc[2], false, false);
        acc[3] = __builtin_amdgcn_wmma_f32_16x16x32_f16(false, a, false, b3, (short)0, acc[3], false, false);

        __syncthreads();
        p ^= 1;
    }

    // Epilogue. C/D layout: VGPR r -> M = r + 8*(lane/16), N = lane%16.
    const float* dv = dinv + (size_t)t * BN;
    const float* mk = mask + (size_t)t * BN;
    #pragma unroll
    for (int nt = 0; nt < 4; ++nt) {
        int h = nt * 16 + l16;
        #pragma unroll
        for (int r = 0; r < 8; ++r) {
            int i = i0 + wave * 16 + half * 8 + r;
            if (i < BN) {
                float v = dv[i] * acc[nt][r];
                if (EPI == 0) v = fmaxf(v + bias[h], 0.0f);
                else          v = (v + bias[h]) * mk[i];
                out[((size_t)t * BN + i) * H + h] = v;
            }
        }
    }
}

// ---------------- c[t] = sum(h2[t]) / (n_act[t]*H) ----------------
__global__ void k_c(const float* __restrict__ h2, const float* __restrict__ mask,
                    float* __restrict__ c) {
    __shared__ float red[256];
    int t = blockIdx.x, tid = threadIdx.x;
    const float* hp = h2 + (size_t)t * BN * H;
    float s = 0.0f;
    for (int e = tid; e < BN * H; e += 256) s += hp[e];
    red[tid] = s;
    __syncthreads();
    for (int w = 128; w > 0; w >>= 1) { if (tid < w) red[tid] += red[tid + w]; __syncthreads(); }
    float total = red[0];
    __syncthreads();
    float nm = 0.0f;
    for (int e = tid; e < BN; e += 256) nm += mask[(size_t)t * BN + e];
    red[tid] = nm;
    __syncthreads();
    for (int w = 128; w > 0; w >>= 1) { if (tid < w) red[tid] += red[tid + w]; __syncthreads(); }
    if (tid == 0) {
        float na = red[0];
        c[t] = (na > 0.0f) ? total / (na * (float)H) : 0.0f;
    }
}

// ---------------- SE gate ----------------
__global__ void k_s(const float* __restrict__ c, const float* __restrict__ Wse1,
                    const float* __restrict__ bse1, const float* __restrict__ Wse2,
                    const float* __restrict__ bse2, float* __restrict__ s) {
    __shared__ float cb[T];
    __shared__ float rb[TSQ];
    int tid = threadIdx.x;
    if (tid < T) cb[tid] = c[tid];
    __syncthreads();
    if (tid < TSQ) {
        float a = bse1[tid];
        for (int t = 0; t < T; ++t) a += cb[t] * Wse1[t * TSQ + tid];
        rb[tid] = fmaxf(a, 0.0f);
    }
    __syncthreads();
    if (tid < T) {
        float a = bse2[tid];
        for (int q = 0; q < TSQ; ++q) a += rb[q] * Wse2[q * T + tid];
        s[tid] = 1.0f / (1.0f + expf(-a));
    }
}

// ---------------- out[b,n,t,:] = (sum_t s[t]*h2[t,j]) @ Wo + bo ----------------
__global__ void k_out(const float* __restrict__ h2, const float* __restrict__ s,
                      const float* __restrict__ Wo, const float* __restrict__ bo,
                      float* __restrict__ out) {
    __shared__ float Z[H];
    __shared__ float sv[T];
    int j = blockIdx.x, tid = threadIdx.x;   // 64 threads
    if (tid < T) sv[tid] = s[tid];
    __syncthreads();
    float z = 0.0f;
    #pragma unroll
    for (int t = 0; t < T; ++t) z += sv[t] * h2[((size_t)t * BN + j) * H + tid];
    Z[tid] = z;
    __syncthreads();
    if (tid < O) {
        float a = bo[tid];
        #pragma unroll
        for (int h = 0; h < H; ++h) a += Z[h] * Wo[h * O + tid];
        for (int t = 0; t < T; ++t)
            out[((size_t)j * T + t) * O + tid] = a;
    }
}

extern "C" void kernel_launch(void* const* d_in, const int* in_sizes, int n_in,
                              void* d_out, int out_size, void* d_ws, size_t ws_size,
                              hipStream_t stream) {
    const float* x    = (const float*)d_in[0];
    const int*   adj  = (const int*)  d_in[1];
    const int*   ego  = (const int*)  d_in[2];
    const float* W1   = (const float*)d_in[3];
    const float* b1   = (const float*)d_in[4];
    const float* W2   = (const float*)d_in[5];
    const float* b2   = (const float*)d_in[6];
    const float* Wse1 = (const float*)d_in[7];
    const float* bse1 = (const float*)d_in[8];
    const float* Wse2 = (const float*)d_in[9];
    const float* bse2 = (const float*)d_in[10];
    const float* Wo   = (const float*)d_in[11];
    const float* bo   = (const float*)d_in[12];
    float* out = (float*)d_out;

    char* w = (char*)d_ws;
    float*    mask   = (float*)(w + OFF_MASK);
    float*    dinv   = (float*)(w + OFF_DINV);
    float*    colsum = (float*)(w + OFF_COLSUM);
    float*    cbuf   = (float*)(w + OFF_C);
    float*    sbuf   = (float*)(w + OFF_S);
    _Float16* Y      = (_Float16*)(w + OFF_Y);
    float*    h1     = (float*)(w + OFF_H1);
    float*    h2     = (float*)(w + OFF_H2);

    const int g_tbn  = (T * BN + 255) / 256;
    const int g_tbnh = (T * BN * H + 255) / 256;

    k_mask<<<g_tbn, 256, 0, stream>>>(ego, mask, colsum);
    dim3 gc((BN + 255) / 256, T, JSPLIT);
    k_colsum<<<gc, 256, 0, stream>>>(adj, mask, colsum);
    k_dinv<<<g_tbn, 256, 0, stream>>>(mask, colsum, dinv);

    k_y1<<<g_tbnh, 256, 0, stream>>>(x, W1, dinv, Y);
    dim3 gp((BN + MT - 1) / MT, T);
    k_prop<0><<<gp, 256, 0, stream>>>(adj, Y, dinv, mask, b1, h1);
    k_y2<<<g_tbnh, 256, 0, stream>>>(h1, W2, dinv, Y);
    k_prop<1><<<gp, 256, 0, stream>>>(adj, Y, dinv, mask, b2, h2);

    k_c<<<T, 256, 0, stream>>>(h2, mask, cbuf);
    k_s<<<1, 64, 0, stream>>>(cbuf, Wse1, bse1, Wse2, bse2, sbuf);
    k_out<<<BN, 64, 0, stream>>>(h2, sbuf, Wo, bo, out);
}
